// FCGF_spconv_70995809402953
// MI455X (gfx1250) — matrix operand order repaired and checked by reading, not compiled, not run on hardware
//
#include <hip/hip_runtime.h>
#include <hip/hip_bf16.h>

// ---------------------------------------------------------------------------
// CDNA5 (gfx1250) implicit-GEMM 3D conv U-Net:
//   v_wmma_f32_16x16x32_bf16 + TDM tensor_load_to_lds weight staging.
// ---------------------------------------------------------------------------

typedef __attribute__((ext_vector_type(16))) __bf16 v16bf;
typedef __attribute__((ext_vector_type(8)))  __bf16 v8bf;
typedef __attribute__((ext_vector_type(8)))  float  v8f;
typedef __attribute__((ext_vector_type(4)))  unsigned int u32x4;
typedef __attribute__((ext_vector_type(8)))  int i32x8;
typedef __attribute__((ext_vector_type(4)))  int i32x4;

#ifndef __has_builtin
#define __has_builtin(x) 0
#endif
#if __has_builtin(__builtin_amdgcn_tensor_load_to_lds) && __has_builtin(__builtin_amdgcn_s_wait_tensorcnt)
#define HAVE_TDM 1
#else
#define HAVE_TDM 0
#endif

#define V0 (64*64*64)
#define V1 (32*32*32)
#define V2 (16*16*16)
#define V3 (8*8*8)

// ---------------- BN fold: s = g*rsqrt(v+eps), b = beta - mean*s -----------
__global__ void k_fold_bn(const float* __restrict__ g, const float* __restrict__ be,
                          const float* __restrict__ mu, const float* __restrict__ va,
                          float* __restrict__ s, float* __restrict__ b, int C) {
    int c = blockIdx.x * blockDim.x + threadIdx.x;
    if (c >= C) return;
    float sc = g[c] * rsqrtf(va[c] + 1e-3f);
    s[c] = sc;
    b[c] = be[c] - mu[c] * sc;
}

// ---------------- Weight pack: DHWIO f32 -> WMMA B-tile bf16 ---------------
// elem = ((((pair*taps + tap)*nCh + chunk)*2 + sub)*32 + lane)*16 + j
// lane<16: N=lane, K={0..7,16..23}[j] ; lane>=16: N=lane-16, K={8..15,24..31}[j]  (K within 32-chunk)
// A (pair,tap) block is contiguous: nCh*2*512 bf16 = nCh*2048 bytes (TDM unit).
__global__ void k_pack_w(const float* __restrict__ w, __bf16* __restrict__ pk,
                         int taps, int Cin, int Cout) {
    int total = taps * Cin * Cout;
    int idx = blockIdx.x * blockDim.x + threadIdx.x;
    if (idx >= total) return;
    int j    = idx & 15;
    int lane = (idx >> 4) & 31;
    int sub  = (idx >> 9) & 1;
    int rest = idx >> 10;
    int nCh   = Cin >> 5;
    int chunk = rest % nCh;
    int rest2 = rest / nCh;
    int tap   = rest2 % taps;
    int pair  = rest2 / taps;
    int cin  = chunk * 32 + ((lane >= 16) ? 8 : 0) + ((j >= 8) ? 8 : 0) + j;
    int cout = (pair * 2 + sub) * 16 + (lane & 15);
    pk[idx] = (__bf16)w[((size_t)tap * Cin + cin) * Cout + cout];
}

// ---------------- conv1: 5x5x5, Cin=1 -> Cout=32, BN + mask, bf16 out ------
__global__ void k_conv1(const float* __restrict__ x, const float* __restrict__ w,
                        const float* __restrict__ s, const float* __restrict__ b,
                        const float* __restrict__ mask, __bf16* __restrict__ out) {
    int vox = blockIdx.x * blockDim.x + threadIdx.x;
    if (vox >= V0) return;
    int xi = vox & 63, t = vox >> 6, yi = t & 63, zi = t >> 6;
    float acc[32];
#pragma unroll
    for (int c = 0; c < 32; ++c) acc[c] = 0.f;
    for (int kz = 0; kz < 5; ++kz) {
        int iz = zi + kz - 2; if ((unsigned)iz >= 64u) continue;
        for (int ky = 0; ky < 5; ++ky) {
            int iy = yi + ky - 2; if ((unsigned)iy >= 64u) continue;
            for (int kx = 0; kx < 5; ++kx) {
                int ix = xi + kx - 2; if ((unsigned)ix >= 64u) continue;
                float xv = x[(iz * 64 + iy) * 64 + ix];
                const float* wp = &w[((kz * 5 + ky) * 5 + kx) * 32];
#pragma unroll
                for (int c = 0; c < 32; ++c) acc[c] += xv * wp[c];
            }
        }
    }
    float mk = mask[vox];
#pragma unroll
    for (int c = 0; c < 32; ++c)
        out[(size_t)vox * 32 + c] = (__bf16)((acc[c] * s[c] + b[c]) * mk);
}

// ---------------- mask downsample: 3^3 any-pool, stride 2, pad 1 -----------
__global__ void k_downmask(const float* __restrict__ mi, float* __restrict__ mo, int od) {
    int n = od * od * od;
    int idx = blockIdx.x * blockDim.x + threadIdx.x;
    if (idx >= n) return;
    int id = od * 2;
    int x = idx % od, t = idx / od, y = t % od, z = t / od;
    float a = 0.f;
    for (int kz = 0; kz < 3; ++kz) {
        int iz = 2 * z + kz - 1; if ((unsigned)iz >= (unsigned)id) continue;
        for (int ky = 0; ky < 3; ++ky) {
            int iy = 2 * y + ky - 1; if ((unsigned)iy >= (unsigned)id) continue;
            for (int kx = 0; kx < 3; ++kx) {
                int ix = 2 * x + kx - 1; if ((unsigned)ix >= (unsigned)id) continue;
                a += mi[((size_t)iz * id + iy) * id + ix];
            }
        }
    }
    mo[idx] = (a > 0.f) ? 1.f : 0.f;
}

// ---------------- channel-slice copy (for skip concats) --------------------
__global__ void k_copych(const __bf16* __restrict__ src, int srcStride,
                         __bf16* __restrict__ dst, int dstStride, int dstOff,
                         int C, int nVox) {
    int idx = blockIdx.x * blockDim.x + threadIdx.x;
    if (idx >= nVox * C) return;
    int c = idx % C, v = idx / C;
    dst[(size_t)v * dstStride + dstOff + c] = src[(size_t)v * srcStride + c];
}

// ---------------- L2 normalize over 32 ch, * mask0, f32 out ----------------
__global__ void k_norm(const __bf16* __restrict__ u, const float* __restrict__ mask,
                       float* __restrict__ out) {
    int vox = blockIdx.x * blockDim.x + threadIdx.x;
    if (vox >= V0) return;
    float vals[32], ss = 0.f;
#pragma unroll
    for (int c = 0; c < 32; ++c) { float f = (float)u[(size_t)vox * 32 + c]; vals[c] = f; ss += f * f; }
    float inv = 1.f / fmaxf(sqrtf(ss), 1e-12f);
    float mk = mask[vox];
#pragma unroll
    for (int c = 0; c < 32; ++c) out[(size_t)vox * 32 + c] = vals[c] * inv * mk;
}

// ---------------- generic WMMA implicit-GEMM conv --------------------------
// MODE 0: 3^3 stride-1, pad 1 | 1: 3^3 stride-2 | 2: 3^3 transposed (lhs_dil=2,
// pad (1,2)) | 3: 1x1x1.  NCH = Cin/32.
// Block = 128 threads (4 waves). All waves share one cout pair (N=32); the
// per-tap weight block (NCH*2KB) is TDM-DMA'd into LDS once per block.
// Each wave: 16 output voxels x 32 output channels -> 2 WMMAs per K-chunk.
template <int MODE, int NCH>
__global__ void k_conv_wmma(const __bf16* __restrict__ in, const __bf16* __restrict__ pk,
                            __bf16* __restrict__ out,
                            const float* __restrict__ bnS, const float* __restrict__ bnB,
                            const float* __restrict__ mask,
                            const __bf16* __restrict__ res, int resStride,
                            int outStride, int outOff,
                            int dimShift, int inDim, int reluF) {
    constexpr int taps = (MODE == 3) ? 1 : 27;
    constexpr int Cin  = NCH * 32;
    __shared__ __attribute__((aligned(16))) __bf16 ldsB[NCH * 1024];

    const int outDim = 1 << dimShift;
    const int dimM   = outDim - 1;
    int lane  = threadIdx.x & 31;
    int wave  = threadIdx.x >> 5;
    int mtile = blockIdx.x * 4 + wave;
    int pair  = blockIdx.y;                 // cout pair: channels [pair*32, pair*32+32)
    int mrow  = lane & 15;                  // A-matrix row (M) owned by this lane
    int vox   = mtile * 16 + mrow;
    int ox = vox & dimM;
    int oy = (vox >> dimShift) & dimM;
    int oz = vox >> (2 * dimShift);
    int cb0 = (lane < 16) ? 0 : 8;          // A K-halves: {0..7,16..23} / {8..15,24..31}

    v8f acc0 = {0.f, 0.f, 0.f, 0.f, 0.f, 0.f, 0.f, 0.f};
    v8f acc1 = {0.f, 0.f, 0.f, 0.f, 0.f, 0.f, 0.f, 0.f};
    v8bf zero;
#pragma unroll
    for (int j = 0; j < 8; ++j) zero[j] = (__bf16)0.0f;

    for (int tap = 0; tap < taps; ++tap) {
        // ---- per-lane A address for this tap ----
        int kx = tap % 3, ky = (tap / 3) % 3, kz = tap / 9;
        int iz, iy, ix;
        bool valid = true;
        if (MODE == 0)      { iz = oz + kz - 1;   iy = oy + ky - 1;   ix = ox + kx - 1; }
        else if (MODE == 1) { iz = 2*oz + kz - 1; iy = 2*oy + ky - 1; ix = 2*ox + kx - 1; }
        else if (MODE == 2) {
            valid = (((oz + kz) & 1) != 0) && (((oy + ky) & 1) != 0) && (((ox + kx) & 1) != 0);
            iz = (oz + kz - 1) >> 1; iy = (oy + ky - 1) >> 1; ix = (ox + kx - 1) >> 1;
        } else              { iz = oz; iy = oy; ix = ox; }
        valid = valid && ((unsigned)iz < (unsigned)inDim) &&
                         ((unsigned)iy < (unsigned)inDim) &&
                         ((unsigned)ix < (unsigned)inDim);
        const __bf16* abase = in + (size_t)(((iz * inDim) + iy) * inDim + ix) * Cin;
        if (valid) __builtin_prefetch(abase, 0, 0);   // global_prefetch while weights DMA

        // ---- stage this tap's weight block (NCH*2KB) into LDS ----
        const __bf16* wsrc = pk + (size_t)(pair * taps + tap) * (NCH * 1024);
#if HAVE_TDM
        if (threadIdx.x < 32) {               // one TDM op per block (wave 0)
            unsigned ndw = NCH * 512u;        // dwords in block (data_size = 4B)
            unsigned long long ga = (unsigned long long)(size_t)wsrc;
            unsigned lds_b = (unsigned)(size_t)(&ldsB[0]);
            u32x4 g0 = { 1u,                                   // count=1, user desc
                         lds_b,                                // lds_addr
                         (unsigned)(ga & 0xFFFFFFFFu),         // global_addr[31:0]
                         (unsigned)((ga >> 32) & 0x01FFFFFFu) | 0x80000000u }; // ga[56:32] | type=2
            i32x8 g1 = { (int)0x20000u,                        // data_size=2 (4B), mask=0
                         (int)((ndw & 0xFFFFu) << 16),         // tensor_dim0 lo16
                         (int)((ndw >> 16) | (1u << 16)),      // tensor_dim0 hi16 | tensor_dim1=1
                         (int)(ndw << 16),                     // tile_dim0 = ndw (1-D tile)
                         0,                                    // tile_dim1/2 unused
                         (int)ndw,                             // tensor_dim0_stride lo32
                         0, 0 };
            i32x4 gz4 = {0, 0, 0, 0};
            i32x8 gz8 = {0, 0, 0, 0, 0, 0, 0, 0};
            __builtin_amdgcn_tensor_load_to_lds(g0, g1, gz4, gz4, gz8, 0);
            __builtin_amdgcn_s_wait_tensorcnt(0);
        }
#else
        {
            const v8bf* srcv = (const v8bf*)wsrc;
            v8bf* dstv = (v8bf*)ldsB;
            for (int i = threadIdx.x; i < NCH * 128; i += 128) dstv[i] = srcv[i];
        }
#endif
        __syncthreads();

        // ---- K chunks: A from global, B from LDS, 2 WMMAs per chunk ----
#pragma unroll
        for (int ch = 0; ch < NCH; ++ch) {
            v8bf a0 = valid ? *(const v8bf*)(abase + ch * 32 + cb0)      : zero;
            v8bf a1 = valid ? *(const v8bf*)(abase + ch * 32 + cb0 + 16) : zero;
            const v8bf* bp0 = (const v8bf*)(ldsB + (ch * 2 + 0) * 512 + lane * 16);
            const v8bf* bp1 = (const v8bf*)(ldsB + (ch * 2 + 1) * 512 + lane * 16);
            v8bf b00 = bp0[0], b01 = bp0[1];
            v8bf b10 = bp1[0], b11 = bp1[1];
            v16bf A, B0, B1;
#pragma unroll
            for (int j = 0; j < 8; ++j) {
                A[j] = a0[j];  A[8 + j] = a1[j];
                B0[j] = b00[j]; B0[8 + j] = b01[j];
                B1[j] = b10[j]; B1[8 + j] = b11[j];
            }
            acc0 = __builtin_amdgcn_wmma_f32_16x16x32_bf16(false, A, false, B0, (short)0, acc0, false, false);
            acc1 = __builtin_amdgcn_wmma_f32_16x16x32_bf16(false, A, false, B1, (short)0, acc1, false, false);
        }
        __syncthreads();   // protect LDS before next tap's DMA
    }

    // ---- epilogue: D layout -> lane L: N=L&15, M=(L>>4)*8+r ----
    int mbase = mtile * 16 + ((lane >> 4) * 8);
#pragma unroll
    for (int sub = 0; sub < 2; ++sub) {
        int nco = (pair * 2 + sub) * 16 + (lane & 15);
        float sN = bnS ? bnS[nco] : 1.0f;
        float bN = bnB ? bnB[nco] : 0.0f;
        const v8f& ac = sub ? acc1 : acc0;
#pragma unroll
        for (int r = 0; r < 8; ++r) {
            int v = mbase + r;
            float f = ac[r] * sN + bN;
            if (res)  f += (float)res[(size_t)v * resStride + nco];
            if (reluF) f = fmaxf(f, 0.f);
            if (mask) f *= mask[v];
            out[(size_t)v * outStride + outOff + nco] = (__bf16)f;
        }
    }
}

// ===========================================================================
extern "C" void kernel_launch(void* const* d_in, const int* in_sizes, int n_in,
                              void* d_out, int out_size, void* d_ws, size_t ws_size,
                              hipStream_t stream) {
    (void)in_sizes; (void)n_in; (void)out_size; (void)ws_size;
    char* ws = (char*)d_ws;
    size_t cur = 0;
    auto alloc = [&](size_t bytes) -> char* {
        size_t o = cur; cur = (cur + bytes + 255) & ~(size_t)255; return ws + o;
    };

    const float* x     = (const float*)d_in[0];
    const float* mask0 = (const float*)d_in[1];

    struct BN { const float* s; const float* b; };
    auto foldbn = [&](int gi, int C) -> BN {
        float* s = (float*)alloc((size_t)C * 4);
        float* b = (float*)alloc((size_t)C * 4);
        k_fold_bn<<<(C + 63) / 64, 64, 0, stream>>>((const float*)d_in[gi], (const float*)d_in[gi + 1],
                                                    (const float*)d_in[gi + 2], (const float*)d_in[gi + 3],
                                                    s, b, C);
        return {s, b};
    };
    auto packw = [&](int gi, int taps, int Cin, int Cout) -> const __bf16* {
        size_t total = (size_t)taps * Cin * Cout;
        __bf16* p = (__bf16*)alloc(total * 2);
        k_pack_w<<<(int)((total + 255) / 256), 256, 0, stream>>>((const float*)d_in[gi], p, taps, Cin, Cout);
        return p;
    };
    auto conv = [&](const __bf16* in, const __bf16* pk, __bf16* out, BN bn,
                    const float* mask, const __bf16* res, int resStride,
                    int Cin, int Cout, int outStride, int outOff,
                    int outDim, int inDim, int mode, int relu) {
        int nVox = outDim * outDim * outDim;
        int sh = (outDim == 64) ? 6 : (outDim == 32) ? 5 : (outDim == 16) ? 4 : 3;
        int nCh = Cin / 32;
        dim3 g(nVox / 64, Cout / 32);
#define LC(M, N) k_conv_wmma<M, N><<<g, 128, 0, stream>>>(in, pk, out, bn.s, bn.b, mask, \
                 res, resStride, outStride, outOff, sh, inDim, relu)
        switch (mode * 16 + nCh) {
            case 0 * 16 + 1: LC(0, 1); break;
            case 0 * 16 + 2: LC(0, 2); break;
            case 0 * 16 + 4: LC(0, 4); break;
            case 0 * 16 + 8: LC(0, 8); break;
            case 1 * 16 + 1: LC(1, 1); break;
            case 1 * 16 + 2: LC(1, 2); break;
            case 1 * 16 + 4: LC(1, 4); break;
            case 2 * 16 + 4: LC(2, 4); break;
            case 2 * 16 + 8: LC(2, 8); break;
            case 3 * 16 + 2: LC(3, 2); break;
            case 3 * 16 + 3: LC(3, 3); break;
            default: break;
        }
#undef LC
    };

    // ---- masks ----
    float* mask1 = (float*)alloc((size_t)V1 * 4);
    float* mask2 = (float*)alloc((size_t)V2 * 4);
    float* mask3 = (float*)alloc((size_t)V3 * 4);
    k_downmask<<<(V1 + 255) / 256, 256, 0, stream>>>(mask0, mask1, 32);
    k_downmask<<<(V2 + 255) / 256, 256, 0, stream>>>(mask1, mask2, 16);
    k_downmask<<<(V3 + 255) / 256, 256, 0, stream>>>(mask2, mask3, 8);

    // ---- BN folds (input indices per setup_inputs flattening order) ----
    BN sb1   = foldbn(3, 32);
    BN b1n1  = foldbn(8, 32),   b1n2  = foldbn(13, 32);
    BN sb2   = foldbn(18, 64);
    BN b2n1  = foldbn(23, 64),  b2n2  = foldbn(28, 64);
    BN sb3   = foldbn(33, 128);
    BN b3n1  = foldbn(38, 128), b3n2  = foldbn(43, 128);
    BN sb4   = foldbn(48, 256);
    BN b4n1  = foldbn(53, 256), b4n2  = foldbn(58, 256);
    BN sb4t  = foldbn(63, 128);
    BN b4tn1 = foldbn(68, 128), b4tn2 = foldbn(73, 128);
    BN sb3t  = foldbn(78, 64);
    BN b3tn1 = foldbn(83, 64),  b3tn2 = foldbn(88, 64);
    BN sb2t  = foldbn(93, 64);
    BN b2tn1 = foldbn(98, 64),  b2tn2 = foldbn(103, 64);
    BN noBN  = {nullptr, nullptr};

    // ---- packed weights ----
    const __bf16* p_b1w1 = packw(7, 27, 32, 32);
    const __bf16* p_b1w2 = packw(12, 27, 32, 32);
    const __bf16* p_c2   = packw(17, 27, 32, 64);
    const __bf16* p_b2w1 = packw(22, 27, 64, 64);
    const __bf16* p_b2w2 = packw(27, 27, 64, 64);
    const __bf16* p_c3   = packw(32, 27, 64, 128);
    const __bf16* p_b3w1 = packw(37, 27, 128, 128);
    const __bf16* p_b3w2 = packw(42, 27, 128, 128);
    const __bf16* p_c4   = packw(47, 27, 128, 256);
    const __bf16* p_b4w1 = packw(52, 27, 256, 256);
    const __bf16* p_b4w2 = packw(57, 27, 256, 256);
    const __bf16* p_c4t  = packw(62, 27, 256, 128);
    const __bf16* p_b4tw1= packw(67, 27, 128, 128);
    const __bf16* p_b4tw2= packw(72, 27, 128, 128);
    const __bf16* p_c3t  = packw(77, 27, 256, 64);
    const __bf16* p_b3tw1= packw(82, 27, 64, 64);
    const __bf16* p_b3tw2= packw(87, 27, 64, 64);
    const __bf16* p_c2t  = packw(92, 27, 128, 64);
    const __bf16* p_b2tw1= packw(97, 27, 64, 64);
    const __bf16* p_b2tw2= packw(102, 27, 64, 64);
    const __bf16* p_c1t  = packw(107, 1, 96, 64);
    const __bf16* p_fin  = packw(108, 1, 64, 32);

    // ---- activation buffers ----
    __bf16* pre   = (__bf16*)alloc((size_t)V0 * 64 * 2);   // shared pre-block / identity buffer
    __bf16* tmp   = (__bf16*)alloc((size_t)V0 * 64 * 2);   // shared intra-block temp
    __bf16* skip1 = (__bf16*)alloc((size_t)V0 * 32 * 2);
    __bf16* skip2 = (__bf16*)alloc((size_t)V1 * 64 * 2);
    __bf16* skip3 = (__bf16*)alloc((size_t)V2 * 128 * 2);
    __bf16* b4out = (__bf16*)alloc((size_t)V3 * 256 * 2);
    __bf16* cat2  = (__bf16*)alloc((size_t)V2 * 256 * 2);
    __bf16* cat1  = (__bf16*)alloc((size_t)V1 * 128 * 2);
    __bf16* cat0  = (__bf16*)alloc((size_t)V0 * 96 * 2);

    // ===================== encoder =====================
    k_conv1<<<V0 / 256, 256, 0, stream>>>(x, (const float*)d_in[2], sb1.s, sb1.b, mask0, pre);
    // block1 @ level0, C=32 -> skip1
    conv(pre, p_b1w1, tmp,   b1n1, mask0, nullptr, 0,  32, 32, 32, 0, 64, 64, 0, 1);
    conv(tmp, p_b1w2, skip1, b1n2, mask0, pre,    32,  32, 32, 32, 0, 64, 64, 0, 1);
    // conv2 stride-2 (32->64)
    conv(skip1, p_c2, pre, sb2, mask1, nullptr, 0,  32, 64, 64, 0, 32, 64, 1, 0);
    conv(pre, p_b2w1, tmp,   b2n1, mask1, nullptr, 0,  64, 64, 64, 0, 32, 32, 0, 1);
    conv(tmp, p_b2w2, skip2, b2n2, mask1, pre,    64,  64, 64, 64, 0, 32, 32, 0, 1);
    // conv3 stride-2 (64->128)
    conv(skip2, p_c3, pre, sb3, mask2, nullptr, 0,  64, 128, 128, 0, 16, 32, 1, 0);
    conv(pre, p_b3w1, tmp,   b3n1, mask2, nullptr, 0, 128, 128, 128, 0, 16, 16, 0, 1);
    conv(tmp, p_b3w2, skip3, b3n2, mask2, pre,   128, 128, 128, 128, 0, 16, 16, 0, 1);
    // conv4 stride-2 (128->256)
    conv(skip3, p_c4, pre, sb4, mask3, nullptr, 0, 128, 256, 256, 0, 8, 16, 1, 0);
    conv(pre, p_b4w1, tmp,   b4n1, mask3, nullptr, 0, 256, 256, 256, 0, 8, 8, 0, 1);
    conv(tmp, p_b4w2, b4out, b4n2, mask3, pre,   256, 256, 256, 256, 0, 8, 8, 0, 1);

    // ===================== decoder =====================
    // conv4_tr (transposed, 256->128)
    conv(b4out, p_c4t, pre, sb4t, mask2, nullptr, 0, 256, 128, 128, 0, 16, 8, 2, 0);
    conv(pre, p_b4tw1, tmp,  b4tn1, mask2, nullptr, 0, 128, 128, 128, 0, 16, 16, 0, 1);
    conv(tmp, p_b4tw2, cat2, b4tn2, mask2, pre,   128, 128, 128, 256, 0, 16, 16, 0, 1);
    k_copych<<<(V2 * 128 + 255) / 256, 256, 0, stream>>>(skip3, 128, cat2, 256, 128, 128, V2);
    // conv3_tr (transposed, 256->64)
    conv(cat2, p_c3t, pre, sb3t, mask1, nullptr, 0, 256, 64, 64, 0, 32, 16, 2, 0);
    conv(pre, p_b3tw1, tmp,  b3tn1, mask1, nullptr, 0, 64, 64, 64, 0, 32, 32, 0, 1);
    conv(tmp, p_b3tw2, cat1, b3tn2, mask1, pre,    64, 64, 64, 128, 0, 32, 32, 0, 1);
    k_copych<<<(V1 * 64 + 255) / 256, 256, 0, stream>>>(skip2, 64, cat1, 128, 64, 64, V1);
    // conv2_tr (transposed, 128->64)
    conv(cat1, p_c2t, pre, sb2t, mask0, nullptr, 0, 128, 64, 64, 0, 64, 32, 2, 0);
    conv(pre, p_b2tw1, tmp,  b2tn1, mask0, nullptr, 0, 64, 64, 64, 0, 64, 64, 0, 1);
    conv(tmp, p_b2tw2, cat0, b2tn2, mask0, pre,    64, 64, 64, 96, 0, 64, 64, 0, 1);
    k_copych<<<(V0 * 32 + 255) / 256, 256, 0, stream>>>(skip1, 32, cat0, 96, 64, 32, V0);
    // conv1_tr 1x1 (96->64), ReLU, mask0 -> tmp (free now)
    conv(cat0, p_c1t, tmp, noBN, mask0, nullptr, 0, 96, 64, 64, 0, 64, 64, 3, 1);
    // final 1x1 (64->32) -> pre (free now)
    conv(tmp, p_fin, pre, noBN, nullptr, nullptr, 0, 64, 32, 32, 0, 64, 64, 3, 0);
    // L2 normalize * mask0 -> d_out (f32)
    k_norm<<<V0 / 256, 256, 0, stream>>>(pre, mask0, (float*)d_out);
}
